// EGNNBlock_17815524344040
// MI455X (gfx1250) — matrix operand order repaired and checked by reading, not compiled, not run on hardware
//
#include <hip/hip_runtime.h>
#include <hip/hip_bf16.h>

typedef __bf16 bf16x16 __attribute__((ext_vector_type(16)));
typedef __bf16 bf16x8  __attribute__((ext_vector_type(8)));
typedef float  floatx8 __attribute__((ext_vector_type(8)));

#define C_DIM 128
#define TILE 64
#define EINW 544   // 128 h_i | 128 h_j | 1 abs | 8 rbf | 128 ef | 7 pad | 128 m_ji | 16 pad

static __device__ __forceinline__ __bf16 f2bf(float f) {
    unsigned u; __builtin_memcpy(&u, &f, 4);
    unsigned r = u + 0x7FFFu + ((u >> 16) & 1u);     // RNE
    unsigned short h = (unsigned short)(r >> 16);
    __bf16 b; __builtin_memcpy(&b, &h, 2); return b;
}
static __device__ __forceinline__ float bf2f(__bf16 b) {
    unsigned short h; __builtin_memcpy(&h, &b, 2);
    unsigned u = ((unsigned)h) << 16;
    float f; __builtin_memcpy(&f, &u, 4); return f;
}
static __device__ __forceinline__ float silu_f(float x) { return x / (1.0f + __expf(-x)); }
static __device__ __forceinline__ float sigm_f(float x) { return 1.0f / (1.0f + __expf(-x)); }

static __device__ __forceinline__ floatx8 wmma_bf16(bf16x16 a, bf16x16 b, floatx8 c) {
    return __builtin_amdgcn_wmma_f32_16x16x32_bf16(false, a, false, b, (short)0, c, false, false);
}

// A-fragment: lane<16 holds row M=lane, K blocks [k0..k0+7] and [k0+16..k0+23];
// lane>=16 holds row M=lane-16, K blocks [k0+8..k0+15] and [k0+24..k0+31].
static __device__ __forceinline__ bf16x16 load_a_frag(const __bf16* row, int k0, int half8) {
    const __bf16* p = row + k0 + half8;
    bf16x8 lo = *(const bf16x8*)p;
    bf16x8 hi = *(const bf16x8*)(p + 16);
    return __builtin_shufflevector(lo, hi, 0,1,2,3,4,5,6,7,8,9,10,11,12,13,14,15);
}

// Accumulate A(16 rows, KC*32 K) x B(KC*32, 64 cols picked by ch) into acc[4]
static __device__ __forceinline__ void gemm_acc(const __bf16* S, int stride, int colOff,
                                                int KC, const __bf16* Bp, floatx8* acc,
                                                int rowA, int half8, int ch, int lane) {
    for (int kc = 0; kc < KC; ++kc) {
        bf16x16 a = load_a_frag(S + (size_t)rowA * stride + colOff, kc * 32, half8);
#pragma unroll
        for (int n4 = 0; n4 < 4; ++n4) {
            int nt = ch * 4 + n4;
            bf16x16 b = *(const bf16x16*)(Bp + ((size_t)(kc * 8 + nt) * 32 + lane) * 16);
            acc[n4] = wmma_bf16(a, b, acc[n4]);
        }
    }
}

// ---------------- weight packer: B-fragment layout, K padded to KC*32 ------------
// mode 0: row map identity (< Korig). mode 1: phi_b permutation for staged b_in layout.
__global__ __launch_bounds__(256) void pack_b_kernel(const float* __restrict__ W,
                                                     __bf16* __restrict__ dst,
                                                     int Korig, int KC, int mode) {
    int idx = blockIdx.x * 256 + threadIdx.x;
    if (idx >= KC * 256) return;
    int kc = idx >> 8, rem = idx & 255, nt = rem >> 5, lane = rem & 31;
    int n = nt * 16 + (lane & 15);
    __bf16 out[16];
#pragma unroll
    for (int j = 0; j < 16; ++j) {
        int k = kc * 32 + ((lane < 16) ? 0 : 8) + ((j < 8) ? j : 8 + j);
        int kk;
        if (mode == 1) { // staged: [h_i h_j | abs rbf ef | pad7 | m_ji | pad16]
            if (k < 256) kk = k;
            else if (k < 393) kk = k + 128;   // abs,rbf,ef -> orig rows 384..520
            else if (k < 400) kk = -1;
            else if (k < 528) kk = k - 144;   // m_ji -> orig rows 256..383
            else kk = -1;
        } else {
            kk = (k < Korig) ? k : -1;
        }
        out[j] = f2bf(kk >= 0 ? W[(size_t)kk * C_DIM + n] : 0.0f);
    }
    __bf16* d = dst + (size_t)idx * 16;
#pragma unroll
    for (int j = 0; j < 16; ++j) d[j] = out[j];
}

__global__ __launch_bounds__(256) void zero_kernel(float* __restrict__ p, size_t n) {
    size_t i = (size_t)blockIdx.x * 256 + threadIdx.x;
    if (i < n) p[i] = 0.0f;
}

// ---------------- fused edge kernel ----------------------------------------------
__global__ __launch_bounds__(256) void edge_kernel(
    const float* __restrict__ node_feats, const float* __restrict__ coords,
    const float* __restrict__ edge_feats, const int* __restrict__ eidx,
    const float* __restrict__ cutp, const float* __restrict__ z0k,
    const __bf16* __restrict__ pW1e, const float* __restrict__ b1e,
    const __bf16* __restrict__ pW2e, const float* __restrict__ b2e,
    const __bf16* __restrict__ pW1b, const float* __restrict__ b1b,
    const __bf16* __restrict__ pW2b, const float* __restrict__ b2b,
    const __bf16* __restrict__ pW1a, const float* __restrict__ b1a,
    const float* __restrict__ W2a, const float* __restrict__ b2ap,
    const __bf16* __restrict__ pW1x, const float* __restrict__ b1x,
    const float* __restrict__ W2x, const float* __restrict__ b2xp,
    float* __restrict__ m_i, float* __restrict__ dcoord,
    float* __restrict__ out_edge, float* __restrict__ out_att,
    int N, int E) {
    __shared__ __align__(16) __bf16 s_ein[TILE * EINW];
    __shared__ __align__(16) __bf16 s_hid[TILE * C_DIM];
    __shared__ float s_geo[TILE * 4];
    __shared__ float s_red[256];
    __shared__ float s_att[TILE];
    __shared__ float s_xsc[TILE];
    __shared__ int   s_snd[TILE];
    __shared__ int   s_rcv[TILE];

    const int t = threadIdx.x;
    const int lane = t & 31, w = t >> 5;
    const int rs = w >> 1, ch = w & 1;
    const int half8 = (lane >> 4) << 3;
    const int rowA = rs * 16 + (lane & 15);
    const int colb = lane & 15;
    const int eb = blockIdx.x * TILE;

    // phase 1: geometry + rbf + zero pads
    if (t < TILE) {
        int e = eb + t;
        __bf16* row = &s_ein[t * EINW];
        if (e < E) {
            int s = eidx[e], r = eidx[E + e];
            s_snd[t] = s; s_rcv[t] = r;
            float dx = coords[s * 3 + 0] - coords[r * 3 + 0];
            float dy = coords[s * 3 + 1] - coords[r * 3 + 1];
            float dz = coords[s * 3 + 2] - coords[r * 3 + 2];
            float ab = sqrtf(dx * dx + dy * dy + dz * dz);
            s_geo[t * 4 + 0] = dx; s_geo[t * 4 + 1] = dy;
            s_geo[t * 4 + 2] = dz; s_geo[t * 4 + 3] = ab;
            float cut = cutp[0];
            float pref = sqrtf(2.0f / cut) / ab;
            row[256] = f2bf(ab);
#pragma unroll
            for (int k = 0; k < 8; ++k)
                row[257 + k] = f2bf(pref * __sinf(z0k[k] * ab / cut));
            for (int k = 393; k < 416; ++k) row[k] = f2bf(0.0f); // pad + not-yet m_ji
            for (int k = 528; k < 544; ++k) row[k] = f2bf(0.0f);
        } else {
            s_snd[t] = 0; s_rcv[t] = 0;
            s_geo[t * 4 + 3] = 1.0f;
            for (int k = 0; k < EINW; ++k) row[k] = f2bf(0.0f);
        }
    }
    __syncthreads();

    // phase 2: gather h_i, h_j, edge_feats as bf16
    {
        int el = t >> 2, q = t & 3, e = eb + el;
        if (e < E) {
            int s = s_snd[el], r = s_rcv[el];
            __bf16* row = &s_ein[el * EINW];
#pragma unroll 4
            for (int j = 0; j < 32; ++j) {
                int col = q * 32 + j;
                row[col]       = f2bf(node_feats[(size_t)s * C_DIM + col]);
                row[128 + col] = f2bf(node_feats[(size_t)r * C_DIM + col]);
                row[265 + col] = f2bf(edge_feats[(size_t)e * C_DIM + col]);
            }
        }
    }
    __syncthreads();

    floatx8 acc[4];

    // G1: phi_e layer1 (K=416) -> silu -> s_hid
#pragma unroll
    for (int i = 0; i < 4; ++i) for (int r = 0; r < 8; ++r) acc[i][r] = 0.0f;
    gemm_acc(s_ein, EINW, 0, 13, pW1e, acc, rowA, half8, ch, lane);
#pragma unroll
    for (int n4 = 0; n4 < 4; ++n4) {
        int col = ch * 64 + n4 * 16 + colb;
        float bias = b1e[col];
#pragma unroll
        for (int r = 0; r < 8; ++r) {
            int row = rs * 16 + r + half8;
            s_hid[row * C_DIM + col] = f2bf(silu_f(acc[n4][r] + bias));
        }
    }
    __syncthreads();

    // G2: phi_e layer2 -> m_ji into s_ein cols 400..527
#pragma unroll
    for (int i = 0; i < 4; ++i) for (int r = 0; r < 8; ++r) acc[i][r] = 0.0f;
    gemm_acc(s_hid, C_DIM, 0, 4, pW2e, acc, rowA, half8, ch, lane);
#pragma unroll
    for (int n4 = 0; n4 < 4; ++n4) {
        int col = ch * 64 + n4 * 16 + colb;
        float bias = b2e[col];
#pragma unroll
        for (int r = 0; r < 8; ++r) {
            int row = rs * 16 + r + half8;
            s_ein[row * EINW + 400 + col] = f2bf(acc[n4][r] + bias);
        }
    }
    __syncthreads();

    // G3: phi_b layer1 (K=544, permuted weights) -> silu -> s_hid
#pragma unroll
    for (int i = 0; i < 4; ++i) for (int r = 0; r < 8; ++r) acc[i][r] = 0.0f;
    gemm_acc(s_ein, EINW, 0, 17, pW1b, acc, rowA, half8, ch, lane);
#pragma unroll
    for (int n4 = 0; n4 < 4; ++n4) {
        int col = ch * 64 + n4 * 16 + colb;
        float bias = b1b[col];
#pragma unroll
        for (int r = 0; r < 8; ++r) {
            int row = rs * 16 + r + half8;
            s_hid[row * C_DIM + col] = f2bf(silu_f(acc[n4][r] + bias));
        }
    }
    __syncthreads();

    // G4: phi_b layer2 -> edge_feats_out (global)
#pragma unroll
    for (int i = 0; i < 4; ++i) for (int r = 0; r < 8; ++r) acc[i][r] = 0.0f;
    gemm_acc(s_hid, C_DIM, 0, 4, pW2b, acc, rowA, half8, ch, lane);
#pragma unroll
    for (int n4 = 0; n4 < 4; ++n4) {
        int col = ch * 64 + n4 * 16 + colb;
        float bias = b2b[col];
#pragma unroll
        for (int r = 0; r < 8; ++r) {
            int row = rs * 16 + r + half8;
            int eg = eb + row;
            if (eg < E) out_edge[(size_t)eg * C_DIM + col] = acc[n4][r] + bias;
        }
    }
    __syncthreads();

    // G5: attention layer1 on m_ji -> silu -> s_hid
#pragma unroll
    for (int i = 0; i < 4; ++i) for (int r = 0; r < 8; ++r) acc[i][r] = 0.0f;
    gemm_acc(s_ein, EINW, 400, 4, pW1a, acc, rowA, half8, ch, lane);
#pragma unroll
    for (int n4 = 0; n4 < 4; ++n4) {
        int col = ch * 64 + n4 * 16 + colb;
        float bias = b1a[col];
#pragma unroll
        for (int r = 0; r < 8; ++r) {
            int row = rs * 16 + r + half8;
            s_hid[row * C_DIM + col] = f2bf(silu_f(acc[n4][r] + bias));
        }
    }
    __syncthreads();

    // attention head: 128 -> 1, sigmoid
    {
        int el = t >> 2, q = t & 3;
        float part = 0.0f;
#pragma unroll 4
        for (int j = 0; j < 32; ++j) {
            int jj = q * 32 + j;
            part += bf2f(s_hid[el * C_DIM + jj]) * W2a[jj];
        }
        s_red[t] = part;
    }
    __syncthreads();
    if (t < TILE) {
        float a0 = s_red[t * 4] + s_red[t * 4 + 1] + s_red[t * 4 + 2] + s_red[t * 4 + 3] + b2ap[0];
        float att = sigm_f(a0);
        s_att[t] = att;
        int eg = eb + t;
        if (eg < E) out_att[eg] = att;
    }
    __syncthreads();

    // G6: phi_x layer1 on m_ji -> silu -> s_hid
#pragma unroll
    for (int i = 0; i < 4; ++i) for (int r = 0; r < 8; ++r) acc[i][r] = 0.0f;
    gemm_acc(s_ein, EINW, 400, 4, pW1x, acc, rowA, half8, ch, lane);
#pragma unroll
    for (int n4 = 0; n4 < 4; ++n4) {
        int col = ch * 64 + n4 * 16 + colb;
        float bias = b1x[col];
#pragma unroll
        for (int r = 0; r < 8; ++r) {
            int row = rs * 16 + r + half8;
            s_hid[row * C_DIM + col] = f2bf(silu_f(acc[n4][r] + bias));
        }
    }
    __syncthreads();

    // phi_x head: 128 -> 1
    {
        int el = t >> 2, q = t & 3;
        float part = 0.0f;
#pragma unroll 4
        for (int j = 0; j < 32; ++j) {
            int jj = q * 32 + j;
            part += bf2f(s_hid[el * C_DIM + jj]) * W2x[jj];
        }
        s_red[t] = part;
    }
    __syncthreads();
    if (t < TILE)
        s_xsc[t] = s_red[t * 4] + s_red[t * 4 + 1] + s_red[t * 4 + 2] + s_red[t * 4 + 3] + b2xp[0];
    __syncthreads();

    // segment-sum m_ji * att into m_i (recv = edge_index[0] = sender array)
    {
        int el = t >> 2, q = t & 3, e = eb + el;
        if (e < E) {
            int rv = s_snd[el];
            float att = s_att[el];
#pragma unroll 4
            for (int j = 0; j < 32; ++j) {
                int jj = q * 32 + j;
                float mv = bf2f(s_ein[el * EINW + 400 + jj]);
                atomicAdd(&m_i[(size_t)rv * C_DIM + jj], mv * att);
            }
        }
    }
    // coordinate deltas
    if (t < TILE) {
        int e = eb + t;
        if (e < E) {
            int rv = s_snd[t];
            float ab = s_geo[t * 4 + 3];
            float sc = s_xsc[t] / (ab + 1.0f);
            atomicAdd(&dcoord[(size_t)rv * 3 + 0], s_geo[t * 4 + 0] * sc);
            atomicAdd(&dcoord[(size_t)rv * 3 + 1], s_geo[t * 4 + 1] * sc);
            atomicAdd(&dcoord[(size_t)rv * 3 + 2], s_geo[t * 4 + 2] * sc);
        }
    }
}

// ---------------- node kernel: phi_n + residual, coords add ----------------------
__global__ __launch_bounds__(256) void node_kernel(
    const float* __restrict__ node_feats, const float* __restrict__ coords,
    const float* __restrict__ m_i, const float* __restrict__ dcoord,
    const __bf16* __restrict__ pW1n, const float* __restrict__ b1n,
    const __bf16* __restrict__ pW2n, const float* __restrict__ b2n,
    float* __restrict__ out_node, float* __restrict__ out_coords, int N) {
    __shared__ __align__(16) __bf16 s_nin[TILE * 256];
    __shared__ __align__(16) __bf16 s_hid[TILE * C_DIM];

    const int t = threadIdx.x;
    const int lane = t & 31, w = t >> 5;
    const int rs = w >> 1, ch = w & 1;
    const int half8 = (lane >> 4) << 3;
    const int rowA = rs * 16 + (lane & 15);
    const int colb = lane & 15;
    const int nb = blockIdx.x * TILE;

    {
        int el = t >> 2, q = t & 3, n = nb + el;
        __bf16* row = &s_nin[el * 256];
        if (n < N) {
#pragma unroll 4
            for (int j = 0; j < 32; ++j) {
                int col = q * 32 + j;
                row[col]       = f2bf(node_feats[(size_t)n * C_DIM + col]);
                row[128 + col] = f2bf(m_i[(size_t)n * C_DIM + col]);
            }
        } else {
#pragma unroll 4
            for (int j = 0; j < 32; ++j) {
                int col = q * 32 + j;
                row[col] = f2bf(0.0f); row[128 + col] = f2bf(0.0f);
            }
        }
    }
    __syncthreads();

    floatx8 acc[4];
#pragma unroll
    for (int i = 0; i < 4; ++i) for (int r = 0; r < 8; ++r) acc[i][r] = 0.0f;
    gemm_acc(s_nin, 256, 0, 8, pW1n, acc, rowA, half8, ch, lane);
#pragma unroll
    for (int n4 = 0; n4 < 4; ++n4) {
        int col = ch * 64 + n4 * 16 + colb;
        float bias = b1n[col];
#pragma unroll
        for (int r = 0; r < 8; ++r) {
            int row = rs * 16 + r + half8;
            s_hid[row * C_DIM + col] = f2bf(silu_f(acc[n4][r] + bias));
        }
    }
    __syncthreads();

#pragma unroll
    for (int i = 0; i < 4; ++i) for (int r = 0; r < 8; ++r) acc[i][r] = 0.0f;
    gemm_acc(s_hid, C_DIM, 0, 4, pW2n, acc, rowA, half8, ch, lane);
#pragma unroll
    for (int n4 = 0; n4 < 4; ++n4) {
        int col = ch * 64 + n4 * 16 + colb;
        float bias = b2n[col];
#pragma unroll
        for (int r = 0; r < 8; ++r) {
            int row = rs * 16 + r + half8;
            int ng = nb + row;
            if (ng < N)
                out_node[(size_t)ng * C_DIM + col] =
                    acc[n4][r] + bias + node_feats[(size_t)ng * C_DIM + col];
        }
    }

    if (t < TILE) {
        int n = nb + t;
        if (n < N) {
#pragma unroll
            for (int i = 0; i < 3; ++i)
                out_coords[(size_t)n * 3 + i] = coords[(size_t)n * 3 + i] + dcoord[(size_t)n * 3 + i];
        }
    }
}

extern "C" void kernel_launch(void* const* d_in, const int* in_sizes, int n_in,
                              void* d_out, int out_size, void* d_ws, size_t ws_size,
                              hipStream_t stream) {
    const float* node_feats = (const float*)d_in[0];
    const float* coords     = (const float*)d_in[1];
    const float* edge_feats = (const float*)d_in[2];
    const int*   eidx       = (const int*)d_in[3];
    const float* cutp       = (const float*)d_in[4];
    const float* z0k        = (const float*)d_in[5];
    const float* W1e = (const float*)d_in[6],  *b1e = (const float*)d_in[7];
    const float* W2e = (const float*)d_in[8],  *b2e = (const float*)d_in[9];
    const float* W1b = (const float*)d_in[10], *b1b = (const float*)d_in[11];
    const float* W2b = (const float*)d_in[12], *b2b = (const float*)d_in[13];
    const float* W1n = (const float*)d_in[14], *b1n = (const float*)d_in[15];
    const float* W2n = (const float*)d_in[16], *b2n = (const float*)d_in[17];
    const float* W1a = (const float*)d_in[18], *b1a = (const float*)d_in[19];
    const float* W2a = (const float*)d_in[20], *b2a = (const float*)d_in[21];
    const float* W1x = (const float*)d_in[22], *b1x = (const float*)d_in[23];
    const float* W2x = (const float*)d_in[24], *b2x = (const float*)d_in[25];

    const int N = in_sizes[0] / C_DIM;
    const int E = in_sizes[3] / 2;

    // workspace layout
    char* ws = (char*)d_ws;
    size_t off = 0;
    float* m_i    = (float*)(ws + off); off += (size_t)N * C_DIM * sizeof(float);
    float* dcoord = (float*)(ws + off); off += (size_t)N * 3 * sizeof(float);
    off = (off + 255) & ~(size_t)255;
    __bf16* pW1e = (__bf16*)(ws + off); off += (size_t)13 * 4096 * 2;
    __bf16* pW2e = (__bf16*)(ws + off); off += (size_t)4  * 4096 * 2;
    __bf16* pW1b = (__bf16*)(ws + off); off += (size_t)17 * 4096 * 2;
    __bf16* pW2b = (__bf16*)(ws + off); off += (size_t)4  * 4096 * 2;
    __bf16* pW1n = (__bf16*)(ws + off); off += (size_t)8  * 4096 * 2;
    __bf16* pW2n = (__bf16*)(ws + off); off += (size_t)4  * 4096 * 2;
    __bf16* pW1a = (__bf16*)(ws + off); off += (size_t)4  * 4096 * 2;
    __bf16* pW1x = (__bf16*)(ws + off); off += (size_t)4  * 4096 * 2;

    // output layout: node_out | coords_out | edge_out | att
    float* out_node   = (float*)d_out;
    float* out_coords = out_node + (size_t)N * C_DIM;
    float* out_edge   = out_coords + (size_t)N * 3;
    float* out_att    = out_edge + (size_t)E * C_DIM;

    // zero accumulators (m_i and dcoord are contiguous)
    size_t nz = (size_t)N * (C_DIM + 3);
    zero_kernel<<<(unsigned)((nz + 255) / 256), 256, 0, stream>>>(m_i, nz);

    // pack weights to bf16 B-fragment layout
    pack_b_kernel<<<13, 256, 0, stream>>>(W1e, pW1e, 393, 13, 0);
    pack_b_kernel<<<4,  256, 0, stream>>>(W2e, pW2e, 128, 4,  0);
    pack_b_kernel<<<17, 256, 0, stream>>>(W1b, pW1b, 521, 17, 1);
    pack_b_kernel<<<4,  256, 0, stream>>>(W2b, pW2b, 128, 4,  0);
    pack_b_kernel<<<8,  256, 0, stream>>>(W1n, pW1n, 256, 8,  0);
    pack_b_kernel<<<4,  256, 0, stream>>>(W2n, pW2n, 128, 4,  0);
    pack_b_kernel<<<4,  256, 0, stream>>>(W1a, pW1a, 128, 4,  0);
    pack_b_kernel<<<4,  256, 0, stream>>>(W1x, pW1x, 128, 4,  0);

    edge_kernel<<<(E + TILE - 1) / TILE, 256, 0, stream>>>(
        node_feats, coords, edge_feats, eidx, cutp, z0k,
        pW1e, b1e, pW2e, b2e, pW1b, b1b, pW2b, b2b,
        pW1a, b1a, W2a, b2a, pW1x, b1x, W2x, b2x,
        m_i, dcoord, out_edge, out_att, N, E);

    node_kernel<<<(N + TILE - 1) / TILE, 256, 0, stream>>>(
        node_feats, coords, m_i, dcoord, pW1n, b1n, pW2n, b2n,
        out_node, out_coords, N);
}
